// RecurrentStatePredictor_64716567216768
// MI455X (gfx1250) — matrix operand order repaired and checked by reading, not compiled, not run on hardware
//
#include <hip/hip_runtime.h>
#include <hip/hip_bf16.h>
#include <math.h>

// ---------------- model constants ----------------
#define BATCH 32768
#define DM    256
#define NL    4
#define DI    512
#define DS_N  64
#define DC    4
#define DR    16
#define MH    256

// ---------------- tiling -------------------------
#define ROWS    32          // rows per workgroup
#define THREADS 256         // 8 waves (wave32)
#define NWAVE   8

typedef __attribute__((ext_vector_type(16))) _Float16 v16h;
typedef __attribute__((ext_vector_type(8)))  float    v8f;

// ---------------- LDS layout (bytes) -------------
#define LDS_RES   0                               // float [32][256]
#define LDS_HID   (LDS_RES  + ROWS*DM*4)          // float [32][256]
#define LDS_H16   (LDS_HID  + ROWS*DM*4)          // f16   [32][264]
#define H16_STR   264
#define LDS_A     (LDS_H16  + ROWS*H16_STR*2)     // f16   [32][1040]
#define A_STR     1040
#define LDS_B16   (LDS_A    + ROWS*A_STR*2)       // f16   [32][528]
#define B16_STR   528
#define LDS_DBC   (LDS_B16  + ROWS*B16_STR*2)     // float [32][144]
#define DBC_STR   144
#define LDS_DTA   (LDS_DBC  + ROWS*DBC_STR*4)     // f16   [32][40]
#define DTA_STR   40
#define LDS_SCAL  (LDS_DTA  + ROWS*DTA_STR*2)     // float [32]
#define LDS_TOTAL (LDS_SCAL + ROWS*4)             // = 203,904 B (< 320 KB WGP LDS)

// ---------------- ws layout (halves) -------------
#define OFF_IN   ((size_t)0)
#define SZ_IN    ((size_t)NL*2*DI*DM)     // 1048576
#define OFF_XP   (OFF_IN + SZ_IN)
#define SZ_XP    ((size_t)NL*(DR+2*DS_N)*DI) // 294912
#define OFF_DT   (OFF_XP + SZ_XP)
#define SZ_DT    ((size_t)NL*DI*32)       // padded K=16 -> 32
#define OFF_OUT  (OFF_DT + SZ_DT)
#define SZ_OUT   ((size_t)NL*DM*DI)
#define OFF_FC1  (OFF_OUT + SZ_OUT)
#define SZ_FC1   ((size_t)NL*2*MH*DM)
#define OFF_FC2  (OFF_FC1 + SZ_FC1)
#define SZ_FC2   ((size_t)NL*DM*MH)
#define OFF_GATE (OFF_FC2 + SZ_FC2)
#define SZ_GATE  ((size_t)DM*2*DM)
#define OFF_UPD  (OFF_GATE + SZ_GATE)
#define SZ_UPD   ((size_t)DM*DM)
#define WS_HALVES (OFF_UPD + SZ_UPD)

// =================================================
//  weight conversion kernels (f32 -> f16 into d_ws)
// =================================================
__global__ void cvt_f16(const float* __restrict__ s, _Float16* __restrict__ d, int n) {
    int i = blockIdx.x * blockDim.x + threadIdx.x;
    if (i < n) d[i] = (_Float16)s[i];
}

// dt_proj_w [NL*DI][16] -> padded [NL*DI][32] (cols 16..31 = 0)
__global__ void cvt_dt_pad(const float* __restrict__ s, _Float16* __restrict__ d, int n32) {
    int i = blockIdx.x * blockDim.x + threadIdx.x;
    if (i < n32) {
        int r = i >> 5, c = i & 31;
        d[i] = (c < DR) ? (_Float16)s[r * DR + c] : (_Float16)0.f;
    }
}

// =================================================
//  device helpers
// =================================================
__device__ __forceinline__ float wave_red_add(float v) {
    #pragma unroll
    for (int m = 16; m >= 1; m >>= 1) v += __shfl_xor(v, m, 32);
    return v;
}
__device__ __forceinline__ float silu_f(float x)    { return x / (1.f + __expf(-x)); }
__device__ __forceinline__ float sigmoid_f(float x) { return 1.f / (1.f + __expf(-x)); }
__device__ __forceinline__ float softplus_f(float x){ return (x > 20.f) ? x : log1pf(expf(x)); }

// residual = hidden + residual (kept in s_res); dst = rmsnorm(residual)*w in f16
__device__ __forceinline__ void fused_add_rms(float* s_res, const float* s_hid,
                                              _Float16* dst, int dstStride,
                                              const float* __restrict__ w,
                                              int wave, int lane) {
    for (int r = wave; r < ROWS; r += NWAVE) {
        float vals[8];
        float acc = 0.f;
        #pragma unroll
        for (int j = 0; j < 8; j++) {
            int c = lane + 32 * j;
            float x = s_res[r * DM + c] + s_hid[r * DM + c];
            vals[j] = x;
            acc += x * x;
        }
        acc = wave_red_add(acc);
        float inv = rsqrtf(acc * (1.f / DM) + 1e-5f);
        #pragma unroll
        for (int j = 0; j < 8; j++) {
            int c = lane + 32 * j;
            s_res[r * DM + c] = vals[j];
            dst[r * dstStride + c] = (_Float16)(vals[j] * inv * w[c]);
        }
    }
}

// 32xN = 32xK (LDS f16, row-major, stride lda) times W^T, W f16 row-major [N][K].
// WMMA v_wmma_f32_16x16x32_f16; per-lane fragment layouts per CDNA5 ISA 7.12.2.
template <typename Epi>
__device__ __forceinline__ void gemm32(const _Float16* __restrict__ W, int N, int K,
                                       const _Float16* __restrict__ A, int lda,
                                       int wave, int lane, Epi epi) {
    const int mtiles = ROWS / 16;          // 2
    const int ntiles = N / 16;
    const int hi  = lane >> 4;             // 0: lanes 0-15, 1: lanes 16-31
    const int l15 = lane & 15;
    const int ntot = mtiles * ntiles;
    for (int t = wave; t < ntot; t += NWAVE) {
        const int mt = t % mtiles, nt = t / mtiles;
        const int m0 = mt * 16, n0 = nt * 16;
        const int ncol = n0 + l15;
        const _Float16* wrow = W + (size_t)ncol * K + hi * 8;    // B: same k order as A
        const _Float16* arow = A + (size_t)(m0 + l15) * lda + hi * 8;
        // prefetch the weight rows of our next tile (global_prefetch_b8)
        if (t + NWAVE < ntot) {
            int nn0 = ((t + NWAVE) / mtiles) * 16;
            __builtin_prefetch(W + (size_t)(nn0 + l15) * K, 0, 1);
        }
        v8f c = {};
        for (int kb = 0; kb < K; kb += 32) {
            v16h a, b;
            const _Float16* ap = arow + kb;
            const _Float16* bp = wrow + kb;
            #pragma unroll
            for (int i = 0; i < 8; i++) { a[i] = ap[i]; a[i + 8] = ap[i + 16]; }
            #pragma unroll
            for (int i = 0; i < 8; i++) { b[i] = bp[i]; b[i + 8] = bp[i + 16]; }
            c = __builtin_amdgcn_wmma_f32_16x16x32_f16(false, a, false, b,
                                                       (short)0, c, false, false);
        }
        const int rbase = m0 + hi * 8;      // C/D layout: VGPR r -> M = r (+8 hi half)
        #pragma unroll
        for (int r2 = 0; r2 < 8; r2++) epi(rbase + r2, ncol, c[r2]);
    }
}

// =================================================
//  fused model kernel: one 32-row tile per workgroup
// =================================================
__global__ __launch_bounds__(THREADS, 1)
void mamba_fused_kernel(const float* __restrict__ z_t,
                        const float* __restrict__ hidden_state,
                        const float* __restrict__ conv_w,
                        const float* __restrict__ conv_b,
                        const float* __restrict__ dt_proj_b,
                        const float* __restrict__ D_param,
                        const float* __restrict__ norm1_w,
                        const float* __restrict__ norm2_w,
                        const float* __restrict__ norm_f_w,
                        const float* __restrict__ gate_b,
                        const float* __restrict__ upd_b,
                        const float* __restrict__ ln_w,
                        const float* __restrict__ ln_b,
                        const _Float16* __restrict__ w_in,
                        const _Float16* __restrict__ w_xp,
                        const _Float16* __restrict__ w_dt,
                        const _Float16* __restrict__ w_out,
                        const _Float16* __restrict__ w_fc1,
                        const _Float16* __restrict__ w_fc2,
                        const _Float16* __restrict__ w_gate,
                        const _Float16* __restrict__ w_upd,
                        float* __restrict__ out) {
    extern __shared__ char smem[];
    float*    s_res  = (float*)   (smem + LDS_RES);
    float*    s_hid  = (float*)   (smem + LDS_HID);
    _Float16* s_h16  = (_Float16*)(smem + LDS_H16);
    _Float16* s_a    = (_Float16*)(smem + LDS_A);
    _Float16* s_b16  = (_Float16*)(smem + LDS_B16);
    float*    s_dbc  = (float*)   (smem + LDS_DBC);
    _Float16* s_dtA  = (_Float16*)(smem + LDS_DTA);
    float*    s_scal = (float*)   (smem + LDS_SCAL);

    const int tid  = threadIdx.x;
    const int wave = tid >> 5;
    const int lane = tid & 31;
    const int row0 = blockIdx.x * ROWS;

    // init: hidden = z_t ; residual = 0
    for (int i = tid; i < ROWS * DM; i += THREADS) {
        int r = i >> 8, c = i & (DM - 1);
        s_hid[r * DM + c] = z_t[(size_t)(row0 + r) * DM + c];
        s_res[r * DM + c] = 0.f;
    }
    __syncthreads();

    for (int l = 0; l < NL; l++) {
        // residual += hidden ; hidden = rmsnorm(residual, norm1) -> f16
        fused_add_rms(s_res, s_hid, s_h16, H16_STR, norm1_w + l * DM, wave, lane);
        __syncthreads();

        // xz = hidden @ in_proj^T   [32x1024]
        gemm32(w_in + (size_t)l * 2 * DI * DM, 2 * DI, DM, s_h16, H16_STR, wave, lane,
               [&](int r, int c, float v) { s_a[r * A_STR + c] = (_Float16)v; });
        __syncthreads();

        // u = silu(u * conv_w[:, -1] + conv_b)   (in place, cols 0..511)
        for (int i = tid; i < ROWS * DI; i += THREADS) {
            int r = i >> 9, d = i & (DI - 1);
            float u  = (float)s_a[r * A_STR + d];
            float cw = conv_w[((size_t)l * DI + d) * DC + (DC - 1)];
            float ub = u * cw + conv_b[l * DI + d];
            s_a[r * A_STR + d] = (_Float16)silu_f(ub);
        }
        __syncthreads();

        // dbc = u @ x_proj^T   [32x144] f32
        gemm32(w_xp + (size_t)l * (DR + 2 * DS_N) * DI, DR + 2 * DS_N, DI, s_a, A_STR,
               wave, lane,
               [&](int r, int c, float v) { s_dbc[r * DBC_STR + c] = v; });
        __syncthreads();

        // A tile for dt gemm (K padded 16->32) + per-row B.C dot
        for (int i = tid; i < ROWS * 32; i += THREADS) {
            int r = i >> 5, c = i & 31;
            s_dtA[r * DTA_STR + c] = (c < DR) ? (_Float16)s_dbc[r * DBC_STR + c]
                                              : (_Float16)0.f;
        }
        for (int r = wave; r < ROWS; r += NWAVE) {
            const float* dbr = s_dbc + r * DBC_STR;
            float p = dbr[DR + lane]      * dbr[DR + DS_N + lane]
                    + dbr[DR + 32 + lane] * dbr[DR + DS_N + 32 + lane];
            p = wave_red_add(p);
            if (lane == 0) s_scal[r] = p;
        }
        __syncthreads();

        // dt = softplus(dbc[:, :16] @ dt_proj^T + dt_b)   [32x512] f16
        {
            const float* dtb = dt_proj_b + l * DI;
            gemm32(w_dt + (size_t)l * DI * 32, DI, 32, s_dtA, DTA_STR, wave, lane,
                   [&](int r, int c, float v) {
                       s_b16[r * B16_STR + c] = (_Float16)softplus_f(v + dtb[c]);
                   });
        }
        __syncthreads();

        // y = dt*u*(B.C) + D*u ; act = y * silu(z)  (in place over u region)
        for (int i = tid; i < ROWS * DI; i += THREADS) {
            int r = i >> 9, d = i & (DI - 1);
            float u  = (float)s_a[r * A_STR + d];
            float dt = (float)s_b16[r * B16_STR + d];
            float y  = dt * u * s_scal[r] + D_param[l * DI + d] * u;
            float z  = (float)s_a[r * A_STR + DI + d];
            s_a[r * A_STR + d] = (_Float16)(y * silu_f(z));
        }
        __syncthreads();

        // hidden = act @ out_proj^T   [32x256] f32
        gemm32(w_out + (size_t)l * DM * DI, DM, DI, s_a, A_STR, wave, lane,
               [&](int r, int c, float v) { s_hid[r * DM + c] = v; });
        __syncthreads();

        // residual += hidden ; hidden = rmsnorm(residual, norm2)
        fused_add_rms(s_res, s_hid, s_h16, H16_STR, norm2_w + l * DM, wave, lane);
        __syncthreads();

        // fc1: [32x512]
        gemm32(w_fc1 + (size_t)l * 2 * MH * DM, 2 * MH, DM, s_h16, H16_STR, wave, lane,
               [&](int r, int c, float v) { s_a[r * A_STR + c] = (_Float16)v; });
        __syncthreads();

        // m = y1 * silu(g1) -> s_h16
        for (int i = tid; i < ROWS * MH; i += THREADS) {
            int r = i >> 8, c = i & (MH - 1);
            float y1 = (float)s_a[r * A_STR + c];
            float g1 = (float)s_a[r * A_STR + MH + c];
            s_h16[r * H16_STR + c] = (_Float16)(y1 * silu_f(g1));
        }
        __syncthreads();

        // fc2: hidden = m @ fc2^T  [32x256] f32
        gemm32(w_fc2 + (size_t)l * DM * MH, DM, MH, s_h16, H16_STR, wave, lane,
               [&](int r, int c, float v) { s_hid[r * DM + c] = v; });
        __syncthreads();
    }

    // mamba_out = rmsnorm(hidden + residual, norm_f) -> s_h16 (f16)
    fused_add_rms(s_res, s_hid, s_h16, H16_STR, norm_f_w, wave, lane);
    __syncthreads();

    // concat [mamba_out | hidden_state] f16 into s_b16
    for (int i = tid; i < ROWS * DM; i += THREADS) {
        int r = i >> 8, c = i & (DM - 1);
        s_b16[r * B16_STR + c]      = s_h16[r * H16_STR + c];
        s_b16[r * B16_STR + DM + c] =
            (_Float16)hidden_state[(size_t)(row0 + r) * DM + c];
    }
    __syncthreads();

    float* s_f1 = (float*)s_a;                 // gate pre-act [32][256]
    float* s_f2 = ((float*)s_a) + ROWS * DM;   // upd  pre-act [32][256]

    gemm32(w_gate, DM, 2 * DM, s_b16, B16_STR, wave, lane,
           [&](int r, int c, float v) { s_f1[r * DM + c] = v + gate_b[c]; });
    gemm32(w_upd, DM, DM, s_h16, H16_STR, wave, lane,
           [&](int r, int c, float v) { s_f2[r * DM + c] = v + upd_b[c]; });
    __syncthreads();

    // new_hidden = g*tanh(upd) + (1-g)*hidden_state ; store + stage for LN
    float* out_nh = out + (size_t)BATCH * DM;
    for (int i = tid; i < ROWS * DM; i += THREADS) {
        int r = i >> 8, c = i & (DM - 1);
        float g  = sigmoid_f(s_f1[r * DM + c]);
        float u_ = tanhf(s_f2[r * DM + c]);
        float hs = hidden_state[(size_t)(row0 + r) * DM + c];
        float nh = g * u_ + (1.f - g) * hs;
        s_res[r * DM + c] = nh;
        out_nh[(size_t)(row0 + r) * DM + c] = nh;
    }
    __syncthreads();

    // z_next = layernorm(new_hidden) * ln_w + ln_b
    for (int r = wave; r < ROWS; r += NWAVE) {
        float s1 = 0.f, s2 = 0.f;
        #pragma unroll
        for (int j = 0; j < 8; j++) {
            float x = s_res[r * DM + lane + 32 * j];
            s1 += x; s2 += x * x;
        }
        s1 = wave_red_add(s1);
        s2 = wave_red_add(s2);
        float mu  = s1 * (1.f / DM);
        float var = s2 * (1.f / DM) - mu * mu;
        float inv = rsqrtf(var + 1e-5f);
        #pragma unroll
        for (int j = 0; j < 8; j++) {
            int c = lane + 32 * j;
            float x = s_res[r * DM + c];
            out[(size_t)(row0 + r) * DM + c] = (x - mu) * inv * ln_w[c] + ln_b[c];
        }
    }
}

// =================================================
//  launch
// =================================================
extern "C" void kernel_launch(void* const* d_in, const int* in_sizes, int n_in,
                              void* d_out, int out_size, void* d_ws, size_t ws_size,
                              hipStream_t stream) {
    if (ws_size < WS_HALVES * sizeof(_Float16)) return;

    const float* z_t       = (const float*)d_in[0];
    const float* hidden_st = (const float*)d_in[1];
    const float* in_proj_w = (const float*)d_in[2];
    const float* conv_w    = (const float*)d_in[3];
    const float* conv_b    = (const float*)d_in[4];
    const float* x_proj_w  = (const float*)d_in[5];
    const float* dt_proj_w = (const float*)d_in[6];
    const float* dt_proj_b = (const float*)d_in[7];
    // d_in[8] = A_log (unused by the reference math)
    const float* D_param   = (const float*)d_in[9];
    const float* out_proj_w= (const float*)d_in[10];
    const float* norm1_w   = (const float*)d_in[11];
    const float* norm2_w   = (const float*)d_in[12];
    const float* fc1_w     = (const float*)d_in[13];
    const float* fc2_w     = (const float*)d_in[14];
    const float* norm_f_w  = (const float*)d_in[15];
    const float* gate_w    = (const float*)d_in[16];
    const float* gate_b    = (const float*)d_in[17];
    const float* upd_w     = (const float*)d_in[18];
    const float* upd_b     = (const float*)d_in[19];
    const float* ln_w      = (const float*)d_in[20];
    const float* ln_b      = (const float*)d_in[21];

    _Float16* ws = (_Float16*)d_ws;

    auto cvt = [&](const float* src, _Float16* dst, size_t n) {
        cvt_f16<<<(unsigned)((n + 255) / 256), 256, 0, stream>>>(src, dst, (int)n);
    };
    cvt(in_proj_w,  ws + OFF_IN,  SZ_IN);
    cvt(x_proj_w,   ws + OFF_XP,  SZ_XP);
    cvt(out_proj_w, ws + OFF_OUT, SZ_OUT);
    cvt(fc1_w,      ws + OFF_FC1, SZ_FC1);
    cvt(fc2_w,      ws + OFF_FC2, SZ_FC2);
    cvt(gate_w,     ws + OFF_GATE, SZ_GATE);
    cvt(upd_w,      ws + OFF_UPD, SZ_UPD);
    cvt_dt_pad<<<(unsigned)((SZ_DT + 255) / 256), 256, 0, stream>>>(
        dt_proj_w, ws + OFF_DT, (int)SZ_DT);

    dim3 grid(BATCH / ROWS);   // 1024 workgroups x 32 rows
    dim3 block(THREADS);
    mamba_fused_kernel<<<grid, block, LDS_TOTAL, stream>>>(
        z_t, hidden_st, conv_w, conv_b, dt_proj_b, D_param,
        norm1_w, norm2_w, norm_f_w, gate_b, upd_b, ln_w, ln_b,
        ws + OFF_IN, ws + OFF_XP, ws + OFF_DT, ws + OFF_OUT,
        ws + OFF_FC1, ws + OFF_FC2, ws + OFF_GATE, ws + OFF_UPD,
        (float*)d_out);
}
